// NTXentLoss_54743653155063
// MI455X (gfx1250) — matrix operand order repaired and checked by reading, not compiled, not run on hardware
//
#include <hip/hip_runtime.h>

typedef _Float16 v16h __attribute__((ext_vector_type(16)));
typedef _Float16 v8h  __attribute__((ext_vector_type(8)));
typedef float    v8f  __attribute__((ext_vector_type(8)));

constexpr int kD  = 512;     // feature dim
constexpr int kB  = 2048;    // batch (power of 2 -> partner = row ^ kB)
constexpr int kN2 = 4096;    // 2B rows
constexpr int kWavesPerBlk = 4;
constexpr int kTilesPerWave = (kN2 / 16) / kWavesPerBlk;   // 64
constexpr float kInvSqrtTemp = 3.16227766016838f; // 1/sqrt(0.1); folded into zn

// ---------------------------------------------------------------------------
// Kernel 1: row L2-normalize concat(z_i, z_j), scale by 1/sqrt(TEMP), emit f16
// ---------------------------------------------------------------------------
__global__ void ntxent_normalize(const float* __restrict__ z_i,
                                 const float* __restrict__ z_j,
                                 _Float16* __restrict__ zn) {
  __shared__ float red[256];
  const int row = blockIdx.x;
  const int tid = threadIdx.x;
  const float* src = (row < kB) ? (z_i + (size_t)row * kD)
                                : (z_j + (size_t)(row - kB) * kD);
  const float x0 = src[tid];
  const float x1 = src[tid + 256];
  red[tid] = x0 * x0 + x1 * x1;
  __syncthreads();
  for (int s = 128; s > 0; s >>= 1) {
    if (tid < s) red[tid] += red[tid + s];
    __syncthreads();
  }
  const float norm  = sqrtf(red[0]);
  const float scale = kInvSqrtTemp / fmaxf(norm, 1e-8f);
  _Float16* dst = zn + (size_t)row * kD;
  dst[tid]       = (_Float16)(x0 * scale);
  dst[tid + 256] = (_Float16)(x1 * scale);
}

// ---------------------------------------------------------------------------
// Kernel 2: fused  zn @ zn^T  +  softmax denominator + positive gather.
// 4 waves per block share one 16-row M-tile (A in LDS -> registers);
// each wave sweeps every 4th 16-column N-tile, B streamed from L2.
// ---------------------------------------------------------------------------
__global__ void __launch_bounds__(kWavesPerBlk * 32)
ntxent_simloss(const _Float16* __restrict__ zn, float* __restrict__ partials) {
  __shared__ __align__(16) _Float16 aTile[16 * kD];   // 16 KB, shared by 4 waves
  __shared__ float sArr[kWavesPerBlk][16];
  __shared__ float pArr[kWavesPerBlk][16];

  const int tid    = threadIdx.x;
  const int lane   = tid & 31;
  const int waveId = tid >> 5;
  const int mBase  = blockIdx.x * 16;
  const int m      = lane & 15;   // A row within tile / B column within tile
  const int half   = lane >> 4;   // K-half selector per WMMA layout

  // Stage the 16 contiguous A rows (16 KB) into LDS, coalesced, all threads.
  {
    const uint4* gsrc = (const uint4*)(zn + (size_t)mBase * kD);
    uint4* ldst = (uint4*)aTile;
    #pragma unroll
    for (int i = 0; i < 1024 / (kWavesPerBlk * 32); ++i)
      ldst[tid + i * (kWavesPerBlk * 32)] = gsrc[tid + i * (kWavesPerBlk * 32)];
  }
  __syncthreads();

  // A fragments for all 16 K-steps, kept in VGPRs (reused by all N-tiles).
  // A 16x32 as loaded here (not ISA table order): lane%16 = M; lane/16 picks K-half; elements 0..15 at
  // K=base+0..15, base = 16*(lane/16) (matches B's contiguous K order).
  v16h a_frag[16];
  #pragma unroll
  for (int kk = 0; kk < 16; ++kk) {
    const _Float16* ap = aTile + m * kD + kk * 32 + half * 16;
    v8h lo = *(const v8h*)(ap);
    v8h hi = *(const v8h*)(ap + 8);
    a_frag[kk] = __builtin_shufflevector(
        lo, hi, 0, 1, 2, 3, 4, 5, 6, 7, 8, 9, 10, 11, 12, 13, 14, 15);
  }

  float s_run[8], pos[8];
  #pragma unroll
  for (int r = 0; r < 8; ++r) { s_run[r] = 0.0f; pos[r] = 0.0f; }

  // B 32x16 layout: lane%16 = N (a row of zn, since B = zn^T); lane/16 picks
  // K 0..15 vs 16..31 -> 16 contiguous f16 (32 B) per lane per k-step.
  const _Float16* bp0 = zn + (size_t)m * kD + half * 16;
  const int giBase = mBase + 8 * half;

  for (int t = 0; t < kTilesPerWave; ++t) {
    const int jt    = waveId + kWavesPerBlk * t;     // this wave's N-tile
    const int nBase = jt * 16;
    const _Float16* bp = bp0 + (size_t)nBase * kD;
    if (t + 1 < kTilesPerWave)                       // hint next tile into cache
      __builtin_prefetch(bp + (size_t)(kWavesPerBlk * 16) * kD, 0, 1);

    // Preload the entire B tile (32 x b128) so the loads pipeline under WMMA.
    v16h bf[16];
    #pragma unroll
    for (int kk = 0; kk < 16; ++kk)
      bf[kk] = *(const v16h*)(bp + kk * 32);

    v8f c = {};
    #pragma unroll
    for (int kk = 0; kk < 16; ++kk)
      c = __builtin_amdgcn_wmma_f32_16x16x32_f16(
          /*neg_a=*/false, a_frag[kk], /*neg_b=*/false, bf[kk],
          /*c_mod=*/(short)0, c, /*reuse_a=*/false, /*reuse_b=*/false);

    // C/D layout: VGPR r -> row (mBase + r + 8*half), column (nBase + lane%16).
    // Branchless epilogue: selects, no exec-mask churn.
    const int gj = nBase + m;
    #pragma unroll
    for (int r = 0; r < 8; ++r) {
      const int gi  = giBase + r;
      const float v = c[r];
      const float e = __expf(v);                 // logits in [-10,10]: safe
      s_run[r] += (gj == gi) ? 0.0f : e;         // diag == -9e15 fill -> exp = 0
      pos[r]    = (gj == (gi ^ kB)) ? v : pos[r];
    }
  }

  // Row reductions inside each wave: each row lives in one 16-lane half.
  #pragma unroll
  for (int r = 0; r < 8; ++r) {
    float s = s_run[r];
    float p = pos[r];
    #pragma unroll
    for (int off = 1; off < 16; off <<= 1) {
      s += __shfl_xor(s, off, 16);
      p += __shfl_xor(p, off, 16);
    }
    if ((lane & 15) == 0) {                       // lanes 0 and 16
      sArr[waveId][r + 8 * half] = s;
      pArr[waveId][r + 8 * half] = p;
    }
  }
  __syncthreads();

  // Combine the 4 wave-partials per row in fixed order (deterministic).
  if (tid < 16) {
    float st = 0.0f, pt = 0.0f;
    #pragma unroll
    for (int w = 0; w < kWavesPerBlk; ++w) {
      st += sArr[w][tid];
      pt += pArr[w][tid];
    }
    float l = logf(st) - pt;                      // -logp of the positive pair
    #pragma unroll
    for (int off = 1; off < 16; off <<= 1)
      l += __shfl_xor(l, off, 16);
    if (tid == 0) partials[blockIdx.x] = l;
  }
}

// ---------------------------------------------------------------------------
// Kernel 3: deterministic fixed-order final reduction -> scalar loss.
// ---------------------------------------------------------------------------
__global__ void ntxent_finalize(const float* __restrict__ partials,
                                float* __restrict__ out) {
  const int lane = threadIdx.x;
  float t = 0.0f;
  for (int i = lane; i < kN2 / 16; i += 32) t += partials[i];
  #pragma unroll
  for (int off = 1; off < 32; off <<= 1) t += __shfl_xor(t, off, 32);
  if (lane == 0) out[0] = t * (1.0f / (float)kN2);
}

extern "C" void kernel_launch(void* const* d_in, const int* in_sizes, int n_in,
                              void* d_out, int out_size, void* d_ws, size_t ws_size,
                              hipStream_t stream) {
  const float* z_i = (const float*)d_in[0];
  const float* z_j = (const float*)d_in[1];

  _Float16* zn = (_Float16*)d_ws;                                  // 4 MB
  float* partials = (float*)((char*)d_ws +
                             (size_t)kN2 * kD * sizeof(_Float16)); // 1 KB

  ntxent_normalize<<<kN2, 256, 0, stream>>>(z_i, z_j, zn);
  ntxent_simloss<<<kN2 / 16, kWavesPerBlk * 32, 0, stream>>>(zn, partials);
  ntxent_finalize<<<1, 32, 0, stream>>>(partials, (float*)d_out);
}
